// FRAPRQ_59639915872389
// MI455X (gfx1250) — matrix operand order repaired and checked by reading, not compiled, not run on hardware
//
#include <hip/hip_runtime.h>
#include <hip/hip_bf16.h>

// FRAP-style model: prep (history MLP) -> batched WMMA GRU -> phase-relation head.
// GRU is the hot loop: per wave of 16 sequences/step:
//   24x v_wmma_f32_16x16x32_f16 (h @ Whh^T) + 12x v_wmma_f32_16x16x4_f32 (x @ Wih^T)

typedef __attribute__((ext_vector_type(16))) _Float16 v16h;
typedef __attribute__((ext_vector_type(8)))  _Float16 v8h;
typedef __attribute__((ext_vector_type(8)))  float    v8f;
typedef __attribute__((ext_vector_type(2)))  float    v2f;

#define BATCH  2048
#define TSTEPS 64
#define NSEQ   (BATCH*8)   /* 16384 = 2^14 */
#define HDIM   64

__device__ __forceinline__ float sigf(float x){ return 1.0f/(1.0f+__expf(-x)); }

__device__ __forceinline__ v8f splat8(float s){
  v8f v;
  #pragma unroll
  for (int e=0;e<8;++e) v[e]=s;
  return v;
}

__device__ __forceinline__ v8f wmma16(v16h a, v16h b, v8f c){
  return __builtin_amdgcn_wmma_f32_16x16x32_f16(false, a, false, b, (short)0, c, false, false);
}
__device__ __forceinline__ v8f wmma4(v2f a, v2f b, v8f c){
  return __builtin_amdgcn_wmma_f32_16x16x4_f32(false, a, false, b, (short)0, c, false, false);
}

// ---------------------------------------------------------------------------
// Kernel 1: x[t][seq][4] = sigmoid(hist[b,t,{l,l+8,l+16}] @ Wh^T + bh)
// ---------------------------------------------------------------------------
__global__ __launch_bounds__(256) void frap_prep(
    const float* __restrict__ hist, const float* __restrict__ Wh,
    const float* __restrict__ bh, float* __restrict__ x)
{
  int idx = blockIdx.x*256 + threadIdx.x;   // idx = t*NSEQ + seq
  int seq = idx & (NSEQ-1);
  int t   = idx >> 14;
  int b = seq >> 3, l = seq & 7;
  const float* hp = hist + ((size_t)b*TSTEPS + t)*24;
  float c0 = hp[l], c1 = hp[l+8], c2 = hp[l+16];
  float4 o;
  o.x = sigf(c0*Wh[0] + c1*Wh[1]  + c2*Wh[2]  + bh[0]);
  o.y = sigf(c0*Wh[3] + c1*Wh[4]  + c2*Wh[5]  + bh[1]);
  o.z = sigf(c0*Wh[6] + c1*Wh[7]  + c2*Wh[8]  + bh[2]);
  o.w = sigf(c0*Wh[9] + c1*Wh[10] + c2*Wh[11] + bh[3]);
  ((float4*)x)[idx] = o;
}

// ---------------------------------------------------------------------------
// Kernel 2: WMMA GRU. 128 threads = 4 waves, each wave owns 16 sequences.
// D/C layout (16x16 f32, 8 VGPRs): lane<16 -> (M=r, N=lane); lane>=16 -> (M=r+8, N=lane-16)
// Gate tiles: j=0..3 r-gate, 4..7 z-gate, 8..11 n-gate (channel c = 16j + lane%16)
// ---------------------------------------------------------------------------
__global__ __launch_bounds__(128) void frap_gru(
    const float* __restrict__ x, const float* __restrict__ Wih,
    const float* __restrict__ Whh, const float* __restrict__ bih,
    const float* __restrict__ bhh, float* __restrict__ hidden)
{
  __shared__ _Float16 sWhh[192*64];   // f16 copy of Whh, row-major [o][c]
  __shared__ float    sWih[192*4];
  __shared__ _Float16 sH[4][16*64];   // per-wave D->A transpose staging

  const int tid = threadIdx.x;
  for (int i = tid; i < 192*64; i += 128) sWhh[i] = (_Float16)Whh[i];
  for (int i = tid; i < 192*4;  i += 128) sWih[i] = Wih[i];
  __syncthreads();

  const int wave = tid >> 5, l = tid & 31, lm = l & 15;
  const int hi = (l >= 16) ? 1 : 0;
  const int s0 = blockIdx.x*64 + wave*16;
  _Float16* hb = &sH[wave][0];

  // Wih as resident f32 B-fragments (B 4x16: lane<16 holds K0,K1; lane>=16 K2,K3)
  v2f wb[12];
  #pragma unroll
  for (int j = 0; j < 12; ++j){
    const float* p = &sWih[(16*j + lm)*4 + 2*hi];
    v2f w; w.x = p[0]; w.y = p[1]; wb[j] = w;
  }
  // Per-lane bias scalars (channel c = 16j + lm is lane-constant per tile)
  float brz[8], bin_[4], bhn[4];
  #pragma unroll
  for (int j = 0; j < 8; ++j) brz[j] = bih[16*j+lm] + bhh[16*j+lm];
  #pragma unroll
  for (int j = 0; j < 4; ++j){ bin_[j] = bih[128+16*j+lm]; bhn[j] = bhh[128+16*j+lm]; }

  v8f h[4];
  #pragma unroll
  for (int j=0;j<4;++j) h[j] = splat8(0.f);
  v16h hA0 = {}; v16h hA1 = {};        // h as f16 A fragments (two K-tiles of 32)

  for (int t = 0; t < TSTEPS; ++t){
    // x A-fragment (f32 16x4): lane<16 holds K0,K1 of row M=lm; lane>=16 holds K2,K3
    const float* xp = x + (((size_t)t*NSEQ + s0 + lm)<<2) + 2*hi;
    const v2f ax = *(const v2f*)xp;

    v8f acc[8], ghn[4], gin[4];
    #pragma unroll
    for (int j=0;j<8;++j) acc[j] = splat8(brz[j]);
    #pragma unroll
    for (int j=0;j<4;++j){ ghn[j] = splat8(bhn[j]); gin[j] = splat8(bin_[j]); }

    // gh = h @ Whh^T : B frag = 16 contiguous halves Whh[16j+lm][32k + 16*hi ..]
    #pragma unroll
    for (int j=0;j<8;++j){
      const _Float16* wp = &sWhh[(16*j+lm)*64 + 16*hi];
      v16h b0 = *(const v16h*)(wp);
      v16h b1 = *(const v16h*)(wp + 32);
      acc[j] = wmma16(hA0, b0, acc[j]);
      acc[j] = wmma16(hA1, b1, acc[j]);
    }
    #pragma unroll
    for (int j=0;j<4;++j){
      const _Float16* wp = &sWhh[(16*(j+8)+lm)*64 + 16*hi];
      v16h b0 = *(const v16h*)(wp);
      v16h b1 = *(const v16h*)(wp + 32);
      ghn[j] = wmma16(hA0, b0, ghn[j]);
      ghn[j] = wmma16(hA1, b1, ghn[j]);
    }
    // gi = x @ Wih^T : chained into r/z accumulators; n-gate kept separate
    #pragma unroll
    for (int j=0;j<8;++j) acc[j] = wmma4(ax, wb[j], acc[j]);
    #pragma unroll
    for (int j=0;j<4;++j) gin[j] = wmma4(ax, wb[8+j], gin[j]);

    // GRU elementwise update, entirely in D layout
    #pragma unroll
    for (int j=0;j<4;++j){
      #pragma unroll
      for (int r=0;r<8;++r){
        float rg   = sigf(acc[j][r]);
        float zg   = sigf(acc[j+4][r]);
        float cand = tanhf(gin[j][r] + rg*ghn[j][r]);
        h[j][r] = cand + zg*(h[j][r] - cand);
      }
    }

    // D layout -> A fragments via per-wave LDS transpose
    #pragma unroll
    for (int j=0;j<4;++j){
      #pragma unroll
      for (int r=0;r<8;++r){
        hb[(r + 8*hi)*64 + 16*j + lm] = (_Float16)h[j][r];
      }
    }
    asm volatile("s_wait_dscnt 0" ::: "memory");
    {
      // 16-bit A 16x32 (this code's contiguous k order, same as B): lane<16 holds K=0..15, lane>=16 holds K=16..31
      const _Float16* ap = hb + lm*64 + 16*hi;
      v8h a00 = *(const v8h*)(ap);
      v8h a01 = *(const v8h*)(ap + 8);
      v8h a10 = *(const v8h*)(ap + 32);
      v8h a11 = *(const v8h*)(ap + 40);
      #pragma unroll
      for (int e=0;e<8;++e){ hA0[e]=a00[e]; hA0[8+e]=a01[e]; hA1[e]=a10[e]; hA1[8+e]=a11[e]; }
    }
  }

  #pragma unroll
  for (int j=0;j<4;++j){
    #pragma unroll
    for (int r=0;r<8;++r)
      hidden[(size_t)(s0 + r + 8*hi)*64 + 16*j + lm] = h[j][r];
  }
}

// ---------------------------------------------------------------------------
// Kernel 3: per-batch head. 64 threads/block, one block per batch element.
// ---------------------------------------------------------------------------
__device__ const int PLd[16] = {1,3, 0,2, 5,7, 4,6, 0,1, 2,3, 4,5, 6,7};

__global__ __launch_bounds__(64) void frap_head(
  const float* __restrict__ feat, const float* __restrict__ hidden,
  const int* __restrict__ relation, const float* __restrict__ emb_phase,
  const float* __restrict__ Wv, const float* __restrict__ bv,
  const float* __restrict__ Wl, const float* __restrict__ bl,
  const float* __restrict__ emb_const,
  const float* __restrict__ Wcf, const float* __restrict__ bcf,
  const float* __restrict__ Wcc, const float* __restrict__ bcc,
  const float* __restrict__ Wcm, const float* __restrict__ bcm,
  const float* __restrict__ Wfin, const float* __restrict__ bfin,
  float* __restrict__ out)
{
  __shared__ float sWl[16*72];
  __shared__ float sWcf[20*32];
  __shared__ float sWcm[20*20];
  __shared__ float sWcc[20*4];
  __shared__ float sLF[8][72];
  __shared__ float sLine[8][16];
  __shared__ float sPP[8][16];
  __shared__ float sZ[56];

  const int tid = threadIdx.x;
  const int b = blockIdx.x;
  for (int i=tid;i<16*72;i+=64) sWl[i]=Wl[i];
  for (int i=tid;i<20*32;i+=64) sWcf[i]=Wcf[i];
  for (int i=tid;i<20*20;i+=64) sWcm[i]=Wcm[i];
  for (int i=tid;i<20*4; i+=64) sWcc[i]=Wcc[i];
  for (int i=tid;i<512;  i+=64) sLF[i>>6][8+(i&63)] = hidden[(size_t)b*512 + i];
  {
    int l = tid>>3, k = tid&7;
    if (k<4) sLF[l][k] = sigf(feat[b*16+8+l]*Wv[k] + bv[k]);
    else { int pid = (int)(feat[b*16+l]+0.5f); sLF[l][k] = sigf(emb_phase[pid*4 + (k-4)]); }
  }
  __syncthreads();

  for (int i=tid;i<128;i+=64){         // line = relu(lane_feat @ Wl^T + bl)
    int l=i>>4, o=i&15;
    float s = bl[o];
    for (int c=0;c<72;++c) s += sLF[l][c]*sWl[o*72+c];
    sLine[l][o] = fmaxf(s,0.f);
  }
  __syncthreads();
  for (int i=tid;i<128;i+=64){         // pp = line[PHASE_LANES].sum
    int p=i>>4, o=i&15;
    sPP[p][o] = sLine[PLd[2*p]][o] + sLine[PLd[2*p+1]][o];
  }
  __syncthreads();

  if (tid < 56){                       // one (p,q) relation per thread
    int p = tid/7, q = tid%7;
    int jc = q + (q>=p ? 1:0);         // J_IDX[p][q]
    float xf[20];
    #pragma unroll
    for (int o=0;o<20;++o){
      float s = bcf[o];
      for (int c=0;c<16;++c) s += sPP[p][c]*sWcf[o*32+c] + sPP[jc][c]*sWcf[o*32+16+c];
      xf[o] = fmaxf(s,0.f);
    }
    int rel = relation[p*7+q];
    float yx[20];
    #pragma unroll
    for (int o=0;o<20;++o){
      float s = bcc[o];
      for (int k=0;k<4;++k) s += emb_const[rel*4+k]*sWcc[o*4+k];
      yx[o] = fmaxf(s,0.f)*xf[o];
    }
    float zf = bfin[0];
    #pragma unroll
    for (int o=0;o<20;++o){
      float s = bcm[o];
      for (int c=0;c<20;++c) s += yx[c]*sWcm[o*20+c];
      zf += fmaxf(s,0.f)*Wfin[o];
    }
    sZ[tid] = zf;
  }
  __syncthreads();
  if (tid < 8){
    float s = 0.f;
    for (int q=0;q<7;++q) s += sZ[tid*7+q];
    out[(size_t)b*8 + tid] = s;
  }
}

// ---------------------------------------------------------------------------
extern "C" void kernel_launch(void* const* d_in, const int* in_sizes, int n_in,
                              void* d_out, int out_size, void* d_ws, size_t ws_size,
                              hipStream_t stream) {
  const float* feat      = (const float*)d_in[0];
  const float* hist      = (const float*)d_in[1];
  const int*   relation  = (const int*)  d_in[2];
  const float* emb_phase = (const float*)d_in[3];
  const float* Wv        = (const float*)d_in[4];
  const float* bv        = (const float*)d_in[5];
  const float* Wh        = (const float*)d_in[6];
  const float* bh        = (const float*)d_in[7];
  const float* Wih       = (const float*)d_in[8];
  const float* Whh       = (const float*)d_in[9];
  const float* bih       = (const float*)d_in[10];
  const float* bhh       = (const float*)d_in[11];
  const float* Wl        = (const float*)d_in[12];
  const float* bl        = (const float*)d_in[13];
  const float* emb_const = (const float*)d_in[14];
  const float* Wcf       = (const float*)d_in[15];
  const float* bcf       = (const float*)d_in[16];
  const float* Wcc       = (const float*)d_in[17];
  const float* bcc       = (const float*)d_in[18];
  const float* Wcm       = (const float*)d_in[19];
  const float* bcm       = (const float*)d_in[20];
  const float* Wfin      = (const float*)d_in[21];
  const float* bfin      = (const float*)d_in[22];
  float* out = (float*)d_out;

  float* x      = (float*)d_ws;                                       // [T][NSEQ][4] f32
  float* hidden = (float*)((char*)d_ws + (size_t)TSTEPS*NSEQ*4*sizeof(float)); // [NSEQ][64]

  frap_prep<<<(NSEQ*TSTEPS)/256, 256, 0, stream>>>(hist, Wh, bh, x);
  frap_gru <<<NSEQ/64, 128, 0, stream>>>(x, Wih, Whh, bih, bhh, hidden);
  frap_head<<<BATCH, 64, 0, stream>>>(feat, hidden, relation, emb_phase, Wv, bv,
                                      Wl, bl, emb_const, Wcf, bcf, Wcc, bcc,
                                      Wcm, bcm, Wfin, bfin, out);
}
